// CrossDimensionalGraphNet_5806795784577
// MI455X (gfx1250) — matrix operand order repaired and checked by reading, not compiled, not run on hardware
//
// CDNA5 / gfx1250 fused hetero-GAT (Catan board graph), wave32 + WMMA f16.
// v5: kernel templated on dst type; relation bodies instantiated with
// compile-time (prm, srcType) so all workspace offsets are immediates and the
// uniform/scalar pressure (SGPR spills seen in v3/v4) disappears.
#include <hip/hip_runtime.h>

typedef __attribute__((ext_vector_type(16))) _Float16 v16h;
typedef __attribute__((ext_vector_type(2)))  _Float16 v2h;
typedef __attribute__((ext_vector_type(8)))  float    v8f;

#define HEX_N  19
#define VER_N  54
#define ROAD_N 72
#define NSLOT  (HEX_N + VER_N + ROAD_N)  // 145
#define SLOPE  0.2f

// workspace layout (bytes):
//   [0, 3072)      float asv[6][64], adv[6][64]
//   [3072, 3840)   float bias[3][64]   (scale*sum(b_rel) + b_lin per dst type)
//   [4096, 18016)  int   adj[145][3][8] ([0..5]=srcs, [6]=count)
//   [18432, 92160) f16   Wfrag[9][8][32][16]  (frag = ntile*2+kt, WMMA B order;
//                        lin matrices pre-scaled by 1/type_scale)
#define WS_BIAS_F   768
#define WS_ADJ_B    4096
#define WS_W16_B    18432

// ---------------- baked edge lists (COO) ----------------
__device__ static const int H2V_HEX[] = {
0,2,3,0,1,2,0,1,6,0,5,6,0,4,5,0,3,4,1,2,8,1,7,8,1,7,18,1,6,18,2,9,10,2,8,9,
2,3,10,3,10,11,3,4,12,3,11,12,4,5,14,4,13,14,4,12,13,5,6,16,5,15,16,5,14,15,
6,17,18,6,16,17,7,8,7,7,7,18,8,9,8,9,9,9,10,10,10,11,11,11,12,11,12,13,12,13,
14,13,13,14,15,14,15,16,15,15,16,17,16,17,18,17,17,18};
__device__ static const int H2V_VER[] = {
0,0,0,1,1,1,2,2,2,3,3,3,4,4,4,5,5,5,6,6,6,7,7,7,8,8,8,9,9,9,10,10,10,11,11,11,
12,12,12,13,13,13,14,14,14,15,15,15,16,16,16,17,17,17,18,18,18,19,19,19,20,20,20,
21,21,21,22,22,22,23,23,23,24,24,25,26,27,27,28,28,29,30,31,32,32,33,34,34,35,36,
36,37,38,38,39,40,40,41,42,43,43,44,45,45,46,47,48,48,49,50,50,51,52,53};
__device__ static const int R2V_ROAD[] = {
0,1,2,0,3,4,3,5,6,5,7,8,7,9,10,1,9,11,4,12,13,12,14,15,14,16,17,6,16,18,19,20,21,
13,19,22,2,20,23,23,24,25,11,26,27,24,26,28,10,29,30,29,31,32,27,31,33,8,34,35,
34,36,37,30,36,38,18,39,40,35,39,41,15,42,43,42,44,44,45,17,45,46,22,47,48,43,47,
49,50,48,49,21,50,51,51,52,25,52,53,53,54,28,55,56,54,55,33,57,58,56,57,32,59,60,
59,61,58,61,38,62,63,60,62,37,64,65,64,66,63,66,41,67,68,65,67,40,69,70,69,71,68,
71,46,70};
__device__ static const int R2V_VER[] = {
0,0,0,1,1,1,2,2,2,3,3,3,4,4,4,5,5,5,6,6,6,7,7,7,8,8,8,9,9,9,10,10,10,11,11,11,
12,12,12,13,13,13,14,14,14,15,15,15,16,16,16,17,17,17,18,18,18,19,19,19,20,20,20,
21,21,21,22,22,22,23,23,23,24,24,24,25,25,26,26,27,27,27,28,28,28,29,29,30,30,31,
31,32,32,32,33,33,34,34,34,35,35,36,36,36,37,37,38,38,38,39,39,40,40,40,41,41,42,
42,43,43,43,44,44,45,45,45,46,46,47,47,48,48,48,49,49,50,50,50,51,51,52,52,53,53};
__device__ static const int V2V_SRC[] = {
0,0,0,1,1,1,2,2,2,3,3,3,4,4,4,5,5,5,6,6,6,7,7,7,8,8,8,9,9,9,10,10,10,11,11,11,
12,12,12,13,13,13,14,14,14,15,15,15,16,16,16,17,17,17,18,18,18,19,19,19,20,20,20,
21,21,21,22,22,22,23,23,23,24,24,24,25,25,26,26,27,27,27,28,28,28,29,29,30,30,31,
31,32,32,32,33,33,34,34,34,35,35,36,36,36,37,37,38,38,38,39,39,40,40,40,41,41,42,
42,43,43,43,44,44,45,45,45,46,46,47,47,48,48,48,49,49,50,50,50,51,51,52,52,53,53};
__device__ static const int V2V_DST[] = {
1,5,12,0,2,6,1,3,9,2,4,19,3,5,16,0,4,14,1,7,11,6,8,24,7,9,27,2,8,22,11,12,32,
6,10,28,0,10,13,12,15,34,5,15,18,13,14,36,4,17,21,16,18,40,14,17,38,3,20,23,
19,21,45,16,20,43,9,23,50,19,22,48,7,25,29,24,26,25,27,8,26,53,11,29,31,24,28,
31,32,28,30,10,30,33,32,34,13,33,35,34,37,15,37,39,35,36,18,39,42,36,38,17,41,44,
40,42,38,41,21,44,47,40,43,20,46,49,45,47,43,46,23,49,52,45,48,22,51,53,50,52,48,
51,27,50};
__device__ static const int R2R_SRC[] = {
0,1,0,2,1,2,0,3,0,4,3,4,3,5,3,6,5,6,5,7,5,8,7,8,7,9,7,10,9,10,1,9,1,11,9,11,
4,12,4,13,12,13,12,14,12,15,14,15,14,16,14,17,16,17,6,16,6,18,16,18,19,20,19,21,
20,21,13,19,13,22,19,22,2,20,2,23,20,23,23,24,23,25,24,25,11,26,11,27,26,27,24,26,
24,28,26,28,10,29,10,30,29,30,29,31,29,32,31,32,27,31,27,33,31,33,8,34,8,35,34,35,
34,36,34,37,36,37,30,36,30,38,36,38,18,39,18,40,39,40,35,39,35,41,39,41,15,42,15,43,
42,43,42,44,44,45,17,45,17,46,45,46,22,47,22,48,47,48,43,47,49,50,48,49,21,50,21,51,
50,51,51,52,25,52,25,53,52,53,53,54,28,55,28,56,55,56,54,55,33,57,33,58,57,58,56,57,
32,59,32,60,59,60,59,61,58,61,38,62,38,63,62,63,60,62,37,64,37,65,64,65,64,66,63,66,
41,67,41,68,67,68,65,67,40,69,40,70,69,70,69,71,68,71,46,70};
__device__ static const int R2R_DST[] = {
1,0,2,0,2,1,3,0,4,0,4,3,5,3,6,3,6,5,7,5,8,5,8,7,9,7,10,7,10,9,9,1,11,1,11,9,
12,4,13,4,13,12,14,12,15,12,15,14,16,14,17,14,17,16,16,6,18,6,18,16,20,19,21,19,
21,20,19,13,22,13,22,19,20,2,23,2,23,20,24,23,25,23,25,24,26,11,27,11,27,26,26,24,
28,24,28,26,29,10,30,10,30,29,31,29,32,29,32,31,31,27,33,27,33,31,34,8,35,8,35,34,
36,34,37,34,37,36,36,30,38,30,38,36,39,18,40,18,40,39,39,35,41,35,41,39,42,15,43,15,
43,42,44,42,45,44,45,17,46,17,46,45,47,22,48,22,48,47,47,43,50,49,49,48,50,21,51,21,
51,50,52,51,52,25,53,25,53,52,54,53,55,28,56,28,56,55,55,54,57,33,58,33,58,57,57,56,
59,32,60,32,60,59,61,59,61,58,62,38,63,38,63,62,62,60,64,37,65,37,65,64,66,64,66,63,
67,41,68,41,68,67,67,65,69,40,70,40,70,69,71,69,71,68,70,46};

static_assert(sizeof(H2V_HEX)/sizeof(int)  == 114, "H2V src len");
static_assert(sizeof(H2V_VER)/sizeof(int)  == 114, "H2V dst len");
static_assert(sizeof(R2V_ROAD)/sizeof(int) == 144, "R2V src len");
static_assert(sizeof(R2V_VER)/sizeof(int)  == 144, "R2V dst len");
static_assert(sizeof(V2V_SRC)/sizeof(int)  == 144, "V2V src len");
static_assert(sizeof(V2V_DST)/sizeof(int)  == 144, "V2V dst len");
static_assert(sizeof(R2R_SRC)/sizeof(int)  == 252, "R2R src len");
static_assert(sizeof(R2R_DST)/sizeof(int)  == 252, "R2R dst len");

// relation param order (matches params dict): 0=h2v 1=v2h 2=r2v 3=v2r 4=v2v 5=r2r
__device__ static const int* const REL_ESRC[6] = {H2V_HEX, H2V_VER, R2V_ROAD, R2V_VER, V2V_SRC, R2R_SRC};
__device__ static const int* const REL_EDST[6] = {H2V_VER, H2V_HEX, R2V_VER, R2V_ROAD, V2V_DST, R2R_DST};
__device__ static const int REL_NE[6]   = {114, 114, 144, 144, 144, 252};
__device__ static const int TYPE_RELS[3][3] = {{1, -1, -1}, {0, 2, 4}, {3, 5, -1}};
__device__ static const int TYPE_NREL[3]    = {1, 3, 2};

struct Ptrs {
  const float* Ws[6]; const float* Wd[6]; const float* aS[6]; const float* aD[6]; const float* br[6];
  const float* Wlin[3]; const float* blin[3];
};

// A-fragment coordinates: for WMMA f16 A (16x32), row m, K-offset ko:
//   lane = m + 16*((ko>>3)&1),  h = 8*(ko>>4) + (ko&7)
// (inverse of: ko = (lane<16?0:8) + 16*(h>>3) + (h&7))

// ---------------- prep kernel ----------------
__global__ void gat_prep(Ptrs p, void* ws) {
  float*    wf  = (float*)ws;
  int*      adj = (int*)((char*)ws + WS_ADJ_B);
  _Float16* wh  = (_Float16*)((char*)ws + WS_W16_B);
  int t  = blockIdx.x * blockDim.x + threadIdx.x;
  int nt = gridDim.x * blockDim.x;

  // collapsed attention vectors: asv = Ws@as, adv = Wd@ad
  for (int idx = t; idx < 6 * 64; idx += nt) {
    int r = idx >> 6, k = idx & 63;
    float s = 0.f, s2 = 0.f;
    for (int d = 0; d < 64; ++d) {
      s  += p.Ws[r][k * 64 + d] * p.aS[r][d];
      s2 += p.Wd[r][k * 64 + d] * p.aD[r][d];
    }
    wf[idx] = s;
    wf[6 * 64 + idx] = s2;
  }
  // fused per-dst-type bias: scale*sum(b_rel) + b_lin
  for (int idx = t; idx < 3 * 64; idx += nt) {
    int ty = idx >> 6, k = idx & 63;
    float scale = (ty == 0) ? 1.0f : (ty == 1) ? (1.0f / 3.0f) : 0.5f;
    float s = p.blin[ty][k];
    for (int ri = 0; ri < TYPE_NREL[ty]; ++ri) s += scale * p.br[TYPE_RELS[ty][ri]][k];
    wf[WS_BIAS_F + idx] = s;
  }
  // adjacency: for each (slot, ri): up to 6 src ids + count
  for (int idx = t; idx < NSLOT * 3; idx += nt) {
    int slot = idx / 3, ri = idx % 3;
    int type = (slot < HEX_N) ? 0 : (slot < HEX_N + VER_N ? 1 : 2);
    int node = slot - ((type == 0) ? 0 : (type == 1) ? HEX_N : (HEX_N + VER_N));
    int* a = adj + idx * 8;
    if (ri >= TYPE_NREL[type]) { a[6] = 0; continue; }
    int prm = TYPE_RELS[type][ri];
    const int* eS = REL_ESRC[prm];
    const int* eD = REL_EDST[prm];
    const int  ne = REL_NE[prm];
    int c = 0;
    for (int e = 0; e < ne; ++e)
      if (eD[e] == node && c < 6) a[c++] = eS[e];
    a[6] = c;
  }
  // weights -> f16 in WMMA B fragment order: [mat][ntile*2+kt][lane][h].
  // lin matrices (mat 6..8) pre-scaled by 1/type_scale so the single final
  // out = scale*acc + bias is exact.
  for (int idx = t; idx < 9 * 4096; idx += nt) {
    int m  = idx >> 12;
    int f  = (idx >> 9) & 7;
    int ln = (idx >> 4) & 31;
    int h  = idx & 15;
    int ntile = f >> 1, kt = f & 1;
    int nn = ln & 15;
    int kb = (ln < 16) ? 0 : 8;
    int row = kt * 32 + kb + (h & 7) + ((h >> 3) << 4);
    int col = ntile * 16 + nn;
    const float* W = (m < 6) ? p.Ws[m] : p.Wlin[m - 6];
    float mul = (m < 6) ? 1.0f : (m == 6) ? 1.0f : (m == 7) ? 3.0f : 2.0f;
    wh[idx] = (_Float16)(W[row * 64 + col] * mul);
  }
}

// ---------------- WMMA 16x64 @ 64x64, chained register accumulation ----------------
__device__ __forceinline__ void gemm16x64_reg(const _Float16 (*sAf)[32][16],
                                              const _Float16* __restrict__ wfrag,
                                              v8f cacc[4], int lane) {
  v16h a0 = *(const v16h*)&sAf[0][lane][0];   // 32B contiguous -> b128 pair
  v16h a1 = *(const v16h*)&sAf[1][lane][0];
#pragma unroll
  for (int ntile = 0; ntile < 4; ++ntile) {
    v16h b0 = *(const v16h*)(wfrag + (size_t)((ntile * 2 + 0) * 32 + lane) * 16);
    v16h b1 = *(const v16h*)(wfrag + (size_t)((ntile * 2 + 1) * 32 + lane) * 16);
    cacc[ntile] = __builtin_amdgcn_wmma_f32_16x16x32_f16(false, a0, false, b0,
                                                         (short)0, cacc[ntile], false, false);
    cacc[ntile] = __builtin_amdgcn_wmma_f32_16x16x32_f16(false, a1, false, b1,
                                                         (short)0, cacc[ntile], false, false);
  }
}

// write (b, k:even pair) of an A tile into fragment-order LDS as one 32-bit store
__device__ __forceinline__ void storeAfragPair(_Float16 (*sAf)[32][16],
                                               int b, int k, float v0, float v1) {
  int kt = k >> 5, ko = k & 31;
  int ln = b + (((ko >> 3) & 1) << 4);
  int h  = ((ko >> 4) << 3) + (ko & 7);          // even when k is even
  v2h pv; pv[0] = (_Float16)v0; pv[1] = (_Float16)v1;
  *(v2h*)&sAf[kt][ln][h] = pv;
}

// one GAT relation with compile-time descriptor: acc += softmax-combined src @ Ws
template <int PRM, int NSRC, int SLOTBASE, int RI>
__device__ __forceinline__ void gat_relation(
    int lane, int bt, int node, const float* __restrict__ xs,
    const float* __restrict__ ws_f, const int* __restrict__ ws_adj,
    const _Float16* __restrict__ ws_h,
    float (*s_xd)[64], float (*s_xs)[16][64], float (*s_att)[16],
    float* s_adst, _Float16 (*s_Af)[32][16], v8f cacc[4]) {
  const int* a   = ws_adj + (size_t)(((SLOTBASE + node) * 3 + RI) * 8);
  const int ecnt = a[6];                       // uniform; >=1 for every node

  __syncthreads();   // previous GEMM consumed s_Af

  // stage raw src rows (float4, coalesced)
  for (int t = lane; t < ecnt * 16 * 16; t += 32) {
    int e = t >> 8, i = (t >> 4) & 15, q = t & 15;
    ((float4*)&s_xs[e][i][0])[q] =
        ((const float4*)&xs[(size_t)((bt * 16 + i) * NSRC + a[e]) * 64])[q];
  }
  const float4* asv4 = (const float4*)(ws_f + PRM * 64);
  const float4* adv4 = (const float4*)(ws_f + (6 + PRM) * 64);
  __syncthreads();

  // a_dst[b] = x_dst . (Wd@ad)
  if (lane < 16) {
    const float4* xr4 = (const float4*)&s_xd[lane][0];
    float s = 0.f;
#pragma unroll
    for (int q = 0; q < 16; ++q) {
      float4 x4 = xr4[q], w4 = adv4[q];
      s += x4.x * w4.x + x4.y * w4.y + x4.z * w4.z + x4.w * w4.w;
    }
    s_adst[lane] = s;
  }
  __syncthreads();

  // leaky logits per (edge, b):  x_src . (Ws@as) + a_dst
  for (int t = lane; t < ecnt * 16; t += 32) {
    int e = t >> 4, b = t & 15;
    const float4* xr4 = (const float4*)&s_xs[e][b][0];
    float s = 0.f;
#pragma unroll
    for (int q = 0; q < 16; ++q) {
      float4 x4 = xr4[q], w4 = asv4[q];
      s += x4.x * w4.x + x4.y * w4.y + x4.z * w4.z + x4.w * w4.w;
    }
    float l = s + s_adst[b];
    s_att[e][b] = (l > 0.f) ? l : SLOPE * l;
  }
  __syncthreads();

  // softmax over <=6 incoming edges
  if (lane < 16) {
    int b = lane;
    float mx = -1e30f;
    for (int e = 0; e < ecnt; ++e) mx = fmaxf(mx, s_att[e][b]);
    float den = 0.f;
    for (int e = 0; e < ecnt; ++e) den += __expf(s_att[e][b] - mx);
    float inv = 1.0f / den;
    for (int e = 0; e < ecnt; ++e) s_att[e][b] = __expf(s_att[e][b] - mx) * inv;
  }
  __syncthreads();

  // xbar = sum_e alpha_e * x_src[e]  ->  f16 A fragments (packed b32 stores)
  for (int t = lane; t < 16 * 32; t += 32) {
    int b = t >> 5, k = (t & 31) * 2;
    float v0 = 0.f, v1 = 0.f;
    for (int e = 0; e < ecnt; ++e) {
      float al = s_att[e][b];
      v0 += al * s_xs[e][b][k];
      v1 += al * s_xs[e][b][k + 1];
    }
    storeAfragPair(s_Af, b, k, v0, v1);
  }
  __syncthreads();
  gemm16x64_reg(s_Af, ws_h + (size_t)PRM * 4096, cacc, lane);   // acc += xbar @ Ws
}

// ---------------- main fused kernel: one wave per (batch16, dst node) ----------------
// TYPE: 0=hex, 1=vertex, 2=road; grid = (B/16, Ndst)
template <int TYPE>
__global__ __launch_bounds__(32)
__attribute__((amdgpu_waves_per_eu(4)))
void gat_main(const float* __restrict__ xh, const float* __restrict__ xv,
              const float* __restrict__ xr, const void* __restrict__ wsv,
              float* __restrict__ out, int Bsz) {
  constexpr int   Ndst     = (TYPE == 0) ? HEX_N : (TYPE == 1) ? VER_N : ROAD_N;
  constexpr int   SLOTBASE = (TYPE == 0) ? 0 : (TYPE == 1) ? HEX_N : (HEX_N + VER_N);
  constexpr float SCALE    = (TYPE == 0) ? 1.0f : (TYPE == 1) ? (1.0f / 3.0f) : 0.5f;

  const int lane = threadIdx.x;
  const int bt   = blockIdx.x;
  const int node = blockIdx.y;
  const float*    ws_f   = (const float*)wsv;
  const int*      ws_adj = (const int*)((const char*)wsv + WS_ADJ_B);
  const _Float16* ws_h   = (const _Float16*)((const char*)wsv + WS_W16_B);
  const float*    xd     = (TYPE == 0) ? xh : (TYPE == 1) ? xv : xr;

  __shared__ float    s_xd[16][64];
  __shared__ float    s_xs[6][16][64];
  __shared__ float    s_att[6][16];
  __shared__ float    s_adst[16];
  __shared__ _Float16 s_Af[2][32][16];   // WMMA A fragments [kt][lane][h]

  v8f cacc[4];
#pragma unroll
  for (int n = 0; n < 4; ++n) cacc[n] = (v8f){0.f, 0.f, 0.f, 0.f, 0.f, 0.f, 0.f, 0.f};

  // stage x_dst rows (float4)
  for (int t = lane; t < 16 * 16; t += 32) {
    int i = t >> 4, q = t & 15;
    ((float4*)&s_xd[i][0])[q] = ((const float4*)&xd[(size_t)((bt * 16 + i) * Ndst + node) * 64])[q];
  }
  __syncthreads();

  // linear skip first: acc += x_dst @ (Wlin / scale)
  for (int t = lane; t < 16 * 32; t += 32) {
    int b = t >> 5, k = (t & 31) * 2;
    storeAfragPair(s_Af, b, k, s_xd[b][k], s_xd[b][k + 1]);
  }
  __syncthreads();
  gemm16x64_reg(s_Af, ws_h + (size_t)(6 + TYPE) * 4096, cacc, lane);

  // relations, all descriptors compile-time:
  if constexpr (TYPE == 0) {
    gat_relation<1, VER_N,  SLOTBASE, 0>(lane, bt, node, xv, ws_f, ws_adj, ws_h,
                                         s_xd, s_xs, s_att, s_adst, s_Af, cacc);
  } else if constexpr (TYPE == 1) {
    gat_relation<0, HEX_N,  SLOTBASE, 0>(lane, bt, node, xh, ws_f, ws_adj, ws_h,
                                         s_xd, s_xs, s_att, s_adst, s_Af, cacc);
    gat_relation<2, ROAD_N, SLOTBASE, 1>(lane, bt, node, xr, ws_f, ws_adj, ws_h,
                                         s_xd, s_xs, s_att, s_adst, s_Af, cacc);
    gat_relation<4, VER_N,  SLOTBASE, 2>(lane, bt, node, xv, ws_f, ws_adj, ws_h,
                                         s_xd, s_xs, s_att, s_adst, s_Af, cacc);
  } else {
    gat_relation<3, VER_N,  SLOTBASE, 0>(lane, bt, node, xv, ws_f, ws_adj, ws_h,
                                         s_xd, s_xs, s_att, s_adst, s_Af, cacc);
    gat_relation<5, ROAD_N, SLOTBASE, 1>(lane, bt, node, xr, ws_f, ws_adj, ws_h,
                                         s_xd, s_xs, s_att, s_adst, s_Af, cacc);
  }

  // out = SCALE * acc + fused_bias, straight from accumulator registers.
  const size_t obase = (size_t)Bsz * 64 * SLOTBASE;
  const float* bias = ws_f + WS_BIAS_F + TYPE * 64;
  const int nn = lane & 15;
  const int mb = (lane < 16) ? 0 : 8;
#pragma unroll
  for (int ntile = 0; ntile < 4; ++ntile) {
    const int col = ntile * 16 + nn;
    const float bc = bias[col];
#pragma unroll
    for (int i = 0; i < 8; ++i) {
      const int m = mb + i;
      out[obase + (size_t)((bt * 16 + m) * Ndst + node) * 64 + col] =
          SCALE * cacc[ntile][i] + bc;
    }
  }
}

extern "C" void kernel_launch(void* const* d_in, const int* in_sizes, int n_in,
                              void* d_out, int out_size, void* d_ws, size_t ws_size,
                              hipStream_t stream) {
  (void)n_in; (void)out_size; (void)ws_size;
  const float* xh = (const float*)d_in[0];
  const float* xv = (const float*)d_in[1];
  const float* xr = (const float*)d_in[2];
  Ptrs p;
  for (int r = 0; r < 6; ++r) {     // params dict order: h2v,v2h,r2v,v2r,v2v,r2r {Ws,Wd,as,ad,b}
    p.Ws[r] = (const float*)d_in[3 + 5 * r + 0];
    p.Wd[r] = (const float*)d_in[3 + 5 * r + 1];
    p.aS[r] = (const float*)d_in[3 + 5 * r + 2];
    p.aD[r] = (const float*)d_in[3 + 5 * r + 3];
    p.br[r] = (const float*)d_in[3 + 5 * r + 4];
  }
  for (int t = 0; t < 3; ++t) {     // lin_h, lin_v, lin_r {W, b}
    p.Wlin[t] = (const float*)d_in[33 + 2 * t];
    p.blin[t] = (const float*)d_in[33 + 2 * t + 1];
  }
  const int Bsz = in_sizes[0] / (HEX_N * 64);   // 2048
  float* outp = (float*)d_out;
  gat_prep<<<dim3(32), dim3(256), 0, stream>>>(p, d_ws);
  gat_main<0><<<dim3(Bsz / 16, HEX_N),  dim3(32), 0, stream>>>(xh, xv, xr, d_ws, outp, Bsz);
  gat_main<1><<<dim3(Bsz / 16, VER_N),  dim3(32), 0, stream>>>(xh, xv, xr, d_ws, outp, Bsz);
  gat_main<2><<<dim3(Bsz / 16, ROAD_N), dim3(32), 0, stream>>>(xh, xv, xr, d_ws, outp, Bsz);
}
